// HookedAttention_45457933861369
// MI455X (gfx1250) — compile-verified, matrix operand order checked by reading, not run on hardware
//
#include <hip/hip_runtime.h>
#include <hip/hip_bf16.h>

// ---------------------------------------------------------------------------
// HookedAttention for MI455X (gfx1250, wave32, WMMA)
//   B=2, T=2048, D=1024, H=16, DH=64
//   1) qkv   = x @ W_QKV + b_QKV          (f16 intermediate in ws)
//   2) z     = causal attention w/ sink   (flash-style, f16 intermediate)
//   3) out   = z @ W_O + b_O              (f32 output)
// ---------------------------------------------------------------------------

typedef __attribute__((ext_vector_type(16))) _Float16 v16h;
typedef __attribute__((ext_vector_type(8)))  float    v8f;

__device__ __forceinline__ v8f wmma16(v16h a, v16h b, v8f c) {
  // D = A(16x32 f16) * B(32x16 f16) + C(16x16 f32)
  return __builtin_amdgcn_wmma_f32_16x16x32_f16(
      /*neg_a=*/false, a, /*neg_b=*/false, b,
      /*c_mod=*/(short)0, c, /*reuse_a=*/false, /*reuse_b=*/false);
}

// A-fragment (16x32, f16): p points at (row0, k0) of a row-major [M][ldk] array.
// Lane = M row; elems 0..7 -> K = kb+0..7, elems 8..15 -> K = 16+kb+0..7,
// where kb = 8 * (lane >= 16).   (ISA 7.12.2, 16-bit A 16x32)
__device__ __forceinline__ v16h load_afrag(const _Float16* p, int ldk) {
  const int lane = threadIdx.x & 31;
  const int row  = lane & 15;
  const int kb   = (lane >> 4) * 8;
  const _Float16* r = p + (size_t)row * ldk;
  v16h a;
#pragma unroll
  for (int e = 0; e < 8; ++e) a[e]     = r[kb + e];
#pragma unroll
  for (int e = 0; e < 8; ++e) a[8 + e] = r[16 + kb + e];
  return a;
}

// B-fragment (32x16) from memory stored n-major: [n][k] row-major (i.e. B^T in
// memory). Lane = N col; same K mapping as A. Used for K-matrix in Q*K^T.
__device__ __forceinline__ v16h load_bfrag_nmajor(const _Float16* p, int ldk) {
  const int lane = threadIdx.x & 31;
  const int col  = lane & 15;
  const int kb   = (lane >> 4) * 8;
  const _Float16* r = p + (size_t)col * ldk;
  v16h b;
#pragma unroll
  for (int e = 0; e < 8; ++e) b[e]     = r[kb + e];
#pragma unroll
  for (int e = 0; e < 8; ++e) b[8 + e] = r[16 + kb + e];
  return b;
}

// B-fragment (32x16) from memory stored k-major: [k][n] row-major. Lane = N col.
__device__ __forceinline__ v16h load_bfrag_kmajor(const _Float16* p, int ldn) {
  const int lane = threadIdx.x & 31;
  const int col  = lane & 15;
  const int kb   = (lane >> 4) * 8;
  v16h b;
#pragma unroll
  for (int e = 0; e < 8; ++e) b[e]     = p[(size_t)(kb + e) * ldn + col];
#pragma unroll
  for (int e = 0; e < 8; ++e) b[8 + e] = p[(size_t)(16 + kb + e) * ldn + col];
  return b;
}

// ---------------------------------------------------------------------------
// GEMM: C[M,N] = A[M,K] * B[K,N] + bias[N]
//   block tile 64(M) x 128(N) x 32(K), 256 threads = 8 waves (2x4),
//   each wave owns a 32x32 tile = 2x2 WMMA accumulators.
// ---------------------------------------------------------------------------
template <typename AT, typename OT>
__global__ __launch_bounds__(256) void gemm_wmma(
    const AT* __restrict__ A, const float* __restrict__ Bm,
    const float* __restrict__ bias, OT* __restrict__ C,
    int M, int N, int K) {
  __shared__ _Float16 As[64][40];    // padded to dodge bank conflicts
  __shared__ _Float16 Bs[32][136];

  const int tid  = threadIdx.x;
  const int wave = tid >> 5;
  const int wy   = wave >> 2;      // 0..1
  const int wx   = wave & 3;       // 0..3
  const int m0   = blockIdx.y * 64;
  const int n0   = blockIdx.x * 128;

  v8f acc[2][2];
#pragma unroll
  for (int i = 0; i < 2; ++i)
#pragma unroll
    for (int j = 0; j < 2; ++j) acc[i][j] = v8f{0, 0, 0, 0, 0, 0, 0, 0};

  for (int k0 = 0; k0 < K; k0 += 32) {
    {  // stage A tile (64x32), convert to f16
      const int r = tid >> 2;
      const int c = (tid & 3) * 8;
      const AT* src = A + (size_t)(m0 + r) * K + k0 + c;
#pragma unroll
      for (int i = 0; i < 8; ++i) As[r][c + i] = (_Float16)(float)src[i];
    }
    {  // stage B tile (32x128), convert to f16
      const int r = tid >> 3;
      const int c = (tid & 7) * 16;
      const float* src = Bm + (size_t)(k0 + r) * N + n0 + c;
#pragma unroll
      for (int i = 0; i < 16; ++i) Bs[r][c + i] = (_Float16)src[i];
    }
    __syncthreads();

    v16h af[2], bf[2];
    af[0] = load_afrag(&As[wy * 32][0], 40);
    af[1] = load_afrag(&As[wy * 32 + 16][0], 40);
    bf[0] = load_bfrag_kmajor(&Bs[0][wx * 32], 136);
    bf[1] = load_bfrag_kmajor(&Bs[0][wx * 32 + 16], 136);
#pragma unroll
    for (int i = 0; i < 2; ++i)
#pragma unroll
      for (int j = 0; j < 2; ++j) acc[i][j] = wmma16(af[i], bf[j], acc[i][j]);
    __syncthreads();
  }

  // epilogue: bias + store.  D layout: row = elem + 8*(lane>=16), col = lane&15
  const int lane = tid & 31;
  const int colL = lane & 15;
  const int rowH = (lane >> 4) * 8;
#pragma unroll
  for (int i = 0; i < 2; ++i)
#pragma unroll
    for (int j = 0; j < 2; ++j) {
      const int ncol = n0 + wx * 32 + j * 16 + colL;
      const float bv = bias[ncol];
#pragma unroll
      for (int e = 0; e < 8; ++e) {
        const int mrow = m0 + wy * 32 + i * 16 + e + rowH;
        C[(size_t)mrow * N + ncol] = (OT)(acc[i][j][e] + bv);
      }
    }
}

// ---------------------------------------------------------------------------
// Flash attention with sink column.  Grid: (T/64, H, B), block 128 (4 waves).
// Each wave independently handles 16 queries; the sink column is the softmax
// initial state (m0 = sink_logit[h], l0 = 1, contributes nothing to PV).
// ---------------------------------------------------------------------------
__global__ __launch_bounds__(128) void attn_wmma(
    const _Float16* __restrict__ qkv, const float* __restrict__ sink,
    _Float16* __restrict__ z) {
  constexpr int T = 2048, H = 16, DH = 64;
  constexpr int ldq = 3 * H * DH;  // 3072
  constexpr int ldz = H * DH;      // 1024

  const int b    = blockIdx.z;
  const int h    = blockIdx.y;
  const int wave = threadIdx.x >> 5;
  const int lane = threadIdx.x & 31;
  const int qb   = blockIdx.x * 64 + wave * 16;

  const _Float16* qp = qkv + (size_t)b * T * ldq + h * DH;
  const _Float16* kp = qp + H * DH;
  const _Float16* vp = qp + 2 * H * DH;

  __shared__ _Float16 pshared[4][16 * 32];  // per-wave prob transpose buffer
  _Float16* pbuf = pshared[wave];

  // Q: two 16x32 A-frags covering d=0..31, 32..63 (stays in registers)
  v16h qa[2];
  qa[0] = load_afrag(qp + (size_t)qb * ldq, ldq);
  qa[1] = load_afrag(qp + (size_t)qb * ldq + 32, ldq);

  const int colL = lane & 15;
  const int rowH = (lane >> 4) * 8;   // row = e + rowH (matches C/D layout)
  const float scale = 0.125f;         // 1/sqrt(64)

  float m[8], l[8];
  const float sl = sink[h];
#pragma unroll
  for (int e = 0; e < 8; ++e) { m[e] = sl; l[e] = 1.0f; }
  v8f zacc[4];
#pragma unroll
  for (int j = 0; j < 4; ++j) zacc[j] = v8f{0, 0, 0, 0, 0, 0, 0, 0};

  const int kend = (((qb + 15) >> 5) + 1) << 5;  // last causal 32-key tile, excl.
  for (int kt = 0; kt < kend; kt += 32) {
    // ---- scores S[16q x 32k] = Q*K^T * scale, two 16-col subtiles ----
    v8f s[2];
#pragma unroll
    for (int st = 0; st < 2; ++st) {
      const _Float16* kbase = kp + (size_t)(kt + st * 16) * ldq;
      v16h b0 = load_bfrag_nmajor(kbase, ldq);       // d 0..31
      v16h b1 = load_bfrag_nmajor(kbase + 32, ldq);  // d 32..63
      v8f t = v8f{0, 0, 0, 0, 0, 0, 0, 0};
      t = wmma16(qa[0], b0, t);
      t = wmma16(qa[1], b1, t);
      s[st] = t;
    }
    // ---- causal mask + online softmax ----
    float alpha[8];
#pragma unroll
    for (int e = 0; e < 8; ++e) {
      const int row = qb + e + rowH;
#pragma unroll
      for (int st = 0; st < 2; ++st) {
        const int col = kt + st * 16 + colL;
        float v = s[st][e] * scale;
        if (col > row) v = -1e30f;
        s[st][e] = v;
      }
      float mx = fmaxf(s[0][e], s[1][e]);
#pragma unroll
      for (int off = 1; off < 16; off <<= 1)
        mx = fmaxf(mx, __shfl_xor(mx, off, 32));   // reduce within 16-lane half
      const float mn = fmaxf(m[e], mx);
      alpha[e] = __expf(m[e] - mn);
      m[e] = mn;
      const float p0 = __expf(s[0][e] - mn);
      const float p1 = __expf(s[1][e] - mn);
      s[0][e] = p0;
      s[1][e] = p1;
      float ts = p0 + p1;
#pragma unroll
      for (int off = 1; off < 16; off <<= 1) ts += __shfl_xor(ts, off, 32);
      l[e] = l[e] * alpha[e] + ts;
    }
#pragma unroll
    for (int j = 0; j < 4; ++j)
#pragma unroll
      for (int e = 0; e < 8; ++e) zacc[j][e] *= alpha[e];

    // ---- transpose probs to A-frag layout through per-wave LDS ----
#pragma unroll
    for (int st = 0; st < 2; ++st)
#pragma unroll
      for (int e = 0; e < 8; ++e)
        pbuf[(e + rowH) * 32 + st * 16 + colL] = (_Float16)s[st][e];
    // intra-wave LDS RAW: LDS is in-order per wave; fence the scheduler + DScnt
    asm volatile("s_wait_dscnt 0" ::: "memory");
    v16h pa = load_afrag(pbuf, 32);

    // ---- PV: z[16 x 64] += P[16x32] * V[32x64] ----
#pragma unroll
    for (int j = 0; j < 4; ++j) {
      v16h vb = load_bfrag_kmajor(vp + (size_t)kt * ldq + j * 16, ldq);
      zacc[j] = wmma16(pa, vb, zacc[j]);
    }
  }

  // ---- finalize: z = zacc / l, store f16 ----
  _Float16* zp = z + ((size_t)b * T + qb) * ldz + h * DH;
#pragma unroll
  for (int j = 0; j < 4; ++j)
#pragma unroll
    for (int e = 0; e < 8; ++e) {
      const int row = e + rowH;
      zp[(size_t)row * ldz + j * 16 + colL] = (_Float16)(zacc[j][e] / l[e]);
    }
}

// ---------------------------------------------------------------------------
extern "C" void kernel_launch(void* const* d_in, const int* in_sizes, int n_in,
                              void* d_out, int out_size, void* d_ws,
                              size_t ws_size, hipStream_t stream) {
  const float* x      = (const float*)d_in[0];
  const float* W_QKV  = (const float*)d_in[1];
  const float* b_QKV  = (const float*)d_in[2];
  const float* W_O    = (const float*)d_in[3];
  const float* b_O    = (const float*)d_in[4];
  const float* sinkl  = (const float*)d_in[5];
  float* out = (float*)d_out;

  constexpr int Bb = 2, T = 2048, D = 1024, H = 16, DH = 64;
  constexpr int M  = Bb * T;        // 4096
  constexpr int N1 = 3 * H * DH;    // 3072
  constexpr int N2 = D;             // 1024

  const size_t need = ((size_t)M * N1 + (size_t)M * N2) * sizeof(_Float16);
  if (ws_size < need) return;  // scratch too small: nothing safe to do

  _Float16* qkv_h = (_Float16*)d_ws;
  _Float16* z_h   = qkv_h + (size_t)M * N1;

  // 1) qkv = x @ W_QKV + b_QKV   -> f16
  gemm_wmma<float, _Float16><<<dim3(N1 / 128, M / 64), dim3(256), 0, stream>>>(
      x, W_QKV, b_QKV, qkv_h, M, N1, D);

  // 2) flash attention with sink -> z (f16)
  attn_wmma<<<dim3(T / 64, H, Bb), dim3(128), 0, stream>>>(qkv_h, sinkl, z_h);

  // 3) out = z @ W_O + b_O       -> f32
  gemm_wmma<_Float16, float><<<dim3(N2 / 128, M / 64), dim3(256), 0, stream>>>(
      z_h, W_O, b_O, out, M, N2, H * DH);
}
